// TransformerBlock_27238682591854
// MI455X (gfx1250) — matrix operand order repaired and checked by reading, not compiled, not run on hardware
//
#include <hip/hip_runtime.h>
#include <stdint.h>

#define EMB   1024
#define HEADS 16
#define HD    64
#define CTX   2048
#define BATCH 4
#define NROW  (BATCH*CTX)   // 8192
#define FF    (4*EMB)       // 4096

typedef _Float16 h16;
typedef __attribute__((ext_vector_type(8)))  h16   v8h;
typedef __attribute__((ext_vector_type(16))) h16   v16h;
typedef __attribute__((ext_vector_type(8)))  float v8f;

struct Frag16 { union { v16h v; h16   e[16]; }; };
struct FragC  { union { v8f  v; float e[8];  }; };
struct V8H    { union { v8h  v; h16   e[8];  }; };

__device__ inline v8h ldv8(const h16* p){ return *(const v8h*)p; }

__device__ inline Frag16 make_frag(v8h lo, v8h hi){
  Frag16 f;
#pragma unroll
  for (int i = 0; i < 8; ++i){ f.e[i] = lo[i]; f.e[8+i] = hi[i]; }
  return f;
}

// 16x32 f16 A fragment from row-major [rows, ld], top-left (r0, k0).
// Lanes 0-15 hold K 0..15, lanes 16-31 hold K 16..31 (matches B fragment order).
__device__ inline Frag16 loadA_rm(const h16* base, int ld, int r0, int k0, int lane){
  int row   = r0 + (lane & 15);
  int klane = (lane < 16) ? 0 : 16;
  const h16* p = base + (size_t)row*ld + k0 + klane;
  return make_frag(ldv8(p), ldv8(p + 8));
}

// 32x16 f16 B fragment from column-major storage (column n contiguous in k, stride ld).
// Same contiguous k order as loadA_rm: lanes 0-15 hold K 0..15, lanes 16-31 hold K 16..31 of column (lane&15).
__device__ inline Frag16 loadB_cm(const h16* base, int ld, int n0, int k0, int lane){
  int col   = n0 + (lane & 15);
  int kband = (lane < 16) ? 0 : 16;
  const h16* p = base + (size_t)col*ld + k0 + kband;
  return make_frag(ldv8(p), ldv8(p + 8));
}

__device__ inline v8f wmma32(v16h a, v16h b, v8f c){
  return __builtin_amdgcn_wmma_f32_16x16x32_f16(false, a, false, b, (short)0, c, false, false);
}

// CDNA5 async memory->LDS copy (16 bytes per lane), tracked by ASYNCcnt.
__device__ inline void async_g2l_b128(uint32_t lds_byte_addr, const void* gaddr){
  asm volatile("global_load_async_to_lds_b128 %0, %1, off"
               :: "v"(lds_byte_addr), "v"(gaddr) : "memory");
}
__device__ inline void wait_asynccnt0(){
  asm volatile("s_wait_asynccnt 0" ::: "memory");
}

// ---------------------------------------------------------------- converts
__global__ __launch_bounds__(256) void cvt_half(const float* __restrict__ in,
                                                h16* __restrict__ out, int n){
  int i = blockIdx.x * 256 + threadIdx.x;
  if (i < n) out[i] = (h16)in[i];
}

// (HEADS, EMB, HD) x3 -> combined [EMB, 3*EMB] f16 (cols: Q heads | K heads | V heads)
__global__ __launch_bounds__(256) void cvt_qkv(const float* __restrict__ wq,
                                               const float* __restrict__ wk,
                                               const float* __restrict__ wv,
                                               h16* __restrict__ out){
  int idx = blockIdx.x * 256 + threadIdx.x;
  int c  = idx / (3*EMB);
  int j  = idx % (3*EMB);
  int mat = j >> 10, jj = j & 1023;
  int hh = jj >> 6, dd = jj & 63;
  const float* src = (mat == 0) ? wq : (mat == 1) ? wk : wv;
  out[idx] = (h16)src[((size_t)hh*EMB + c)*HD + dd];
}

// ---------------------------------------------------------------- layernorm -> f16
__global__ __launch_bounds__(256) void ln_half(const float* __restrict__ x,
                                               const float* __restrict__ g,
                                               const float* __restrict__ beta,
                                               h16* __restrict__ out){
  const int row = blockIdx.x;
  const int tid = threadIdx.x;
  const float* xr = x + (size_t)row * EMB;
  float vals[4];
  float s = 0.f, ss = 0.f;
#pragma unroll
  for (int j = 0; j < 4; ++j){
    float v = xr[tid + j*256];
    vals[j] = v; s += v; ss += v*v;
  }
  __shared__ float r1[256], r2[256];
  r1[tid] = s; r2[tid] = ss;
  __syncthreads();
  for (int off = 128; off > 0; off >>= 1){
    if (tid < off){ r1[tid] += r1[tid+off]; r2[tid] += r2[tid+off]; }
    __syncthreads();
  }
  float mean = r1[0] * (1.f/EMB);
  float var  = r2[0] * (1.f/EMB) - mean*mean;
  float rstd = rsqrtf(var + 1e-5f);
  h16* orow = out + (size_t)row * EMB;
#pragma unroll
  for (int j = 0; j < 4; ++j){
    int c = tid + j*256;
    orow[c] = (h16)((vals[j]-mean)*rstd*g[c] + beta[c]);
  }
}

// ---------------------------------------------------------------- tiled f16 GEMM
// A tile staged via CDNA5 async global->LDS (no VGPR round-trip); B tile staged
// with a register transpose so both fragment reads are contiguous 16B LDS loads.
// MODE 0: scatter to Q [B,H,T,D], K [B,H,T,D], V^T [B,H,D,T]  (f16)
// MODE 1: outf = acc + bias[col] + resid   (fp32)
// MODE 2: outh = relu(acc + bias[col])     (f16)
template<int MODE>
__global__ __launch_bounds__(256) void gemm_f16(
    const h16* __restrict__ A, const h16* __restrict__ Bm,
    int M, int N, int K,
    const float* __restrict__ bias, const float* __restrict__ resid,
    float* __restrict__ outf, h16* __restrict__ outh,
    h16* __restrict__ qb, h16* __restrict__ kb, h16* __restrict__ vb)
{
  __shared__ h16 As[128*40];   // 128 rows x 32 k (stride 40 halves: bank-friendly)
  __shared__ h16 Bs[128*40];   // 128 cols x 32 k (B staged transposed)
  const int tid  = threadIdx.x;
  const int lane = tid & 31;
  const int wave = tid >> 5;
  const int wrow = wave >> 2;          // 2 wave-rows x 64
  const int wcol = wave & 3;           // 4 wave-cols x 32
  const int rowBase = blockIdx.y * 128;
  const int colBase = blockIdx.x * 128;
  const uint32_t as_base = (uint32_t)(uintptr_t)&As[0];

  FragC acc[4][2];
#pragma unroll
  for (int mi = 0; mi < 4; ++mi)
#pragma unroll
    for (int ni = 0; ni < 2; ++ni)
#pragma unroll
      for (int i = 0; i < 8; ++i) acc[mi][ni].e[i] = 0.f;

  for (int k0 = 0; k0 < K; k0 += 32){
    // stage A tile 128x32: async DMA straight into LDS (ASYNCcnt path)
    for (int c = tid; c < 512; c += 256){
      int ar = c >> 2, ac = (c & 3) << 3;
      async_g2l_b128(as_base + (uint32_t)(ar*40 + ac)*2,
                     A + (size_t)(rowBase+ar)*K + k0 + ac);
    }
    // stage B tile 32x128 transposed -> Bs[n][k]
    for (int c = tid; c < 512; c += 256){
      int bkk = c >> 4, bn = (c & 15) << 3;
      V8H t; t.v = ldv8(Bm + (size_t)(k0+bkk)*N + colBase + bn);
#pragma unroll
      for (int j = 0; j < 8; ++j) Bs[(bn+j)*40 + bkk] = t.e[j];
    }
    if (k0 + 32 < K)   // hint next A tile -> global_prefetch_b8
      __builtin_prefetch(A + (size_t)(rowBase + (tid >> 1))*K + k0 + 32, 0, 0);
    wait_asynccnt0();       // our async LDS writes landed
    __syncthreads();        // everyone's writes visible

    Frag16 af[4], bf[2];
#pragma unroll
    for (int mi = 0; mi < 4; ++mi) af[mi] = loadA_rm(As, 40, wrow*64 + mi*16, 0, lane);
#pragma unroll
    for (int ni = 0; ni < 2; ++ni) bf[ni] = loadB_cm(Bs, 40, wcol*32 + ni*16, 0, lane);
#pragma unroll
    for (int mi = 0; mi < 4; ++mi)
#pragma unroll
      for (int ni = 0; ni < 2; ++ni)
        acc[mi][ni].v = wmma32(af[mi].v, bf[ni].v, acc[mi][ni].v);
    __syncthreads();
  }

  const int nrel = lane & 15;
  const int moff = (lane < 16) ? 0 : 8;
#pragma unroll
  for (int mi = 0; mi < 4; ++mi){
#pragma unroll
    for (int ni = 0; ni < 2; ++ni){
#pragma unroll
      for (int i = 0; i < 8; ++i){
        int grow = rowBase + wrow*64 + mi*16 + moff + i;
        int gcol = colBase + wcol*32 + ni*16 + nrel;
        float val = acc[mi][ni].e[i];
        if (MODE == 0){
          int mat = gcol >> 10, jj = gcol & 1023;
          int hh = jj >> 6, dd = jj & 63;
          int bb = grow >> 11, tt = grow & 2047;
          h16 hv = (h16)val;
          size_t bhh = (size_t)(bb*HEADS + hh);
          if      (mat == 0) qb[(bhh*CTX + tt)*HD + dd] = hv;
          else if (mat == 1) kb[(bhh*CTX + tt)*HD + dd] = hv;
          else               vb[(bhh*HD  + dd)*CTX + tt] = hv;
        } else if (MODE == 1){
          size_t idx = (size_t)grow*N + gcol;
          outf[idx] = val + bias[gcol] + resid[idx];
        } else {
          float r = val + bias[gcol];
          outh[(size_t)grow*N + gcol] = (h16)(r > 0.f ? r : 0.f);
        }
      }
    }
  }
}

// ---------------------------------------------------------------- flash attention
// Per wave: 16 query rows. Computes S^T = K*Q^T (lane = query col) so softmax row
// stats need only a shfl_xor(16); O^T = V^T * P^T accumulated in 4 f32 tiles.
// Full (unmasked) key blocks take a compare-free path; only the diagonal block
// pays for per-element causal predication.
__global__ __launch_bounds__(256) void attn_kernel(
    const h16* __restrict__ Q, const h16* __restrict__ K,
    const h16* __restrict__ Vt, h16* __restrict__ Out)
{
  const int qblocks = CTX/128;
  const int bh   = blockIdx.x / qblocks;
  const int qblk = blockIdx.x % qblocks;
  const int b = bh / HEADS, h = bh % HEADS;
  const int wave = threadIdx.x >> 5;
  const int lane = threadIdx.x & 31;
  const bool lohalf = lane < 16;
  const int q0 = qblk*128 + wave*16;
  const int qcol = q0 + (lane & 15);

  const h16* Qh = Q  + (size_t)bh*CTX*HD;
  const h16* Kh = K  + (size_t)bh*CTX*HD;
  const h16* Vh = Vt + (size_t)bh*HD*CTX;

  Frag16 qb0 = loadB_cm(Qh, HD, q0, 0, lane);   // Q^T, d = 0..31
  Frag16 qb1 = loadB_cm(Qh, HD, q0, 32, lane);  // Q^T, d = 32..63

  float m_run = -1e30f, l_run = 0.f;
  FragC oacc[4];
#pragma unroll
  for (int db = 0; db < 4; ++db)
#pragma unroll
    for (int i = 0; i < 8; ++i) oacc[db].e[i] = 0.f;

  const float scale = 0.125f;   // 1/sqrt(64)

  for (int s0 = 0; s0 <= q0 + 15; s0 += 32){
    FragC clo, chi;
#pragma unroll
    for (int i = 0; i < 8; ++i){ clo.e[i] = 0.f; chi.e[i] = 0.f; }
    {
      Frag16 k00 = loadA_rm(Kh, HD, s0,    0,  lane);
      Frag16 k01 = loadA_rm(Kh, HD, s0,    32, lane);
      clo.v = wmma32(k00.v, qb0.v, clo.v);
      clo.v = wmma32(k01.v, qb1.v, clo.v);
      Frag16 k10 = loadA_rm(Kh, HD, s0+16, 0,  lane);
      Frag16 k11 = loadA_rm(Kh, HD, s0+16, 32, lane);
      chi.v = wmma32(k10.v, qb0.v, chi.v);
      chi.v = wmma32(k11.v, qb1.v, chi.v);
    }
    float plo[8], phi[8];
    float mloc = -1e30f;
    if (s0 + 31 <= q0){
      // entire 32-key block is below the diagonal for every q in this tile
#pragma unroll
      for (int i = 0; i < 8; ++i){
        float v0 = clo.e[i]*scale;
        float v1 = chi.e[i]*scale;
        plo[i] = v0; phi[i] = v1;
        float mm = v0 > v1 ? v0 : v1;
        mloc = mloc > mm ? mloc : mm;
      }
    } else {
      const int sbase = s0 + (lohalf ? 0 : 8);
#pragma unroll
      for (int i = 0; i < 8; ++i){
        int slo = sbase + i, shi = slo + 16;
        float v0 = (slo <= qcol) ? clo.e[i]*scale : -1e30f;   // causal mask
        float v1 = (shi <= qcol) ? chi.e[i]*scale : -1e30f;
        plo[i] = v0; phi[i] = v1;
        float mm = v0 > v1 ? v0 : v1;
        mloc = mloc > mm ? mloc : mm;
      }
    }
    mloc = fmaxf(mloc, __shfl_xor(mloc, 16));
    float m_new = fmaxf(m_run, mloc);
    float corr  = __expf(m_run - m_new);
    float lsum  = 0.f;
#pragma unroll
    for (int i = 0; i < 8; ++i){
      plo[i] = __expf(plo[i] - m_new);
      phi[i] = __expf(phi[i] - m_new);
      lsum  += plo[i] + phi[i];
    }
    lsum += __shfl_xor(lsum, 16);
    l_run = l_run*corr + lsum;
    m_run = m_new;
#pragma unroll
    for (int db = 0; db < 4; ++db)
#pragma unroll
      for (int i = 0; i < 8; ++i) oacc[db].e[i] *= corr;

    // rebuild P^T as a B fragment (half-wave swap via shfl_xor 16)
    float olo[8], ohi[8];
#pragma unroll
    for (int i = 0; i < 8; ++i){
      olo[i] = __shfl_xor(plo[i], 16);
      ohi[i] = __shfl_xor(phi[i], 16);
    }
    Frag16 pb;
#pragma unroll
    for (int i = 0; i < 8; ++i){
      pb.e[i]   = (h16)(lohalf ? plo[i] : ohi[i]);
      pb.e[8+i] = (h16)(lohalf ? olo[i] : phi[i]);
    }
#pragma unroll
    for (int db = 0; db < 4; ++db){
      Frag16 vf = loadA_rm(Vh, CTX, db*16, s0, lane);   // V^T rows d, k = s
      oacc[db].v = wmma32(vf.v, pb.v, oacc[db].v);
    }
  }

  float inv_l = 1.f / l_run;
  size_t outbase = ((size_t)b*CTX + qcol)*EMB + h*HD + (lohalf ? 0 : 8);
#pragma unroll
  for (int db = 0; db < 4; ++db){
    V8H pack;
#pragma unroll
    for (int i = 0; i < 8; ++i) pack.e[i] = (h16)(oacc[db].e[i]*inv_l);
    *(v8h*)(Out + outbase + db*16) = pack.v;
  }
}

// ---------------------------------------------------------------- launch
extern "C" void kernel_launch(void* const* d_in, const int* in_sizes, int n_in,
                              void* d_out, int out_size, void* d_ws, size_t ws_size,
                              hipStream_t stream)
{
  const float* x      = (const float*)d_in[0];
  const float* wq     = (const float*)d_in[1];
  const float* wk     = (const float*)d_in[2];
  const float* wv     = (const float*)d_in[3];
  const float* w_proj = (const float*)d_in[4];
  const float* b_proj = (const float*)d_in[5];
  const float* w1     = (const float*)d_in[6];
  const float* b1     = (const float*)d_in[7];
  const float* w2     = (const float*)d_in[8];
  const float* b2     = (const float*)d_in[9];
  const float* ln1g   = (const float*)d_in[10];
  const float* ln1b   = (const float*)d_in[11];
  const float* ln2g   = (const float*)d_in[12];
  const float* ln2b   = (const float*)d_in[13];
  float* out = (float*)d_out;

  char* ws = (char*)d_ws;
  size_t off = 0;
  auto take = [&](size_t elems) -> h16* {
    h16* p = (h16*)(ws + off);
    off += elems * sizeof(h16);
    off = (off + 255) & ~(size_t)255;
    return p;
  };
  h16* wqkvh  = take((size_t)3*EMB*EMB);
  h16* wprojh = take((size_t)EMB*EMB);
  h16* w1h    = take((size_t)EMB*FF);
  h16* w2h    = take((size_t)FF*EMB);
  h16* hbuf   = take((size_t)NROW*EMB);
  h16* qbuf   = take((size_t)NROW*EMB);
  h16* kbuf   = take((size_t)NROW*EMB);
  h16* vbuf   = take((size_t)NROW*EMB);
  h16* attnb  = take((size_t)NROW*EMB);
  h16* h2buf  = take((size_t)NROW*EMB);
  h16* ff1    = take((size_t)NROW*FF);

  dim3 blk(256);
  cvt_qkv <<<dim3((3*EMB*EMB)/256), blk, 0, stream>>>(wq, wk, wv, wqkvh);
  cvt_half<<<dim3((EMB*EMB)/256),   blk, 0, stream>>>(w_proj, wprojh, EMB*EMB);
  cvt_half<<<dim3((EMB*FF)/256),    blk, 0, stream>>>(w1, w1h, EMB*FF);
  cvt_half<<<dim3((FF*EMB)/256),    blk, 0, stream>>>(w2, w2h, FF*EMB);

  ln_half<<<dim3(NROW), blk, 0, stream>>>(x, ln1g, ln1b, hbuf);

  gemm_f16<0><<<dim3((3*EMB)/128, NROW/128), blk, 0, stream>>>(
      hbuf, wqkvh, NROW, 3*EMB, EMB,
      nullptr, nullptr, nullptr, nullptr, qbuf, kbuf, vbuf);

  attn_kernel<<<dim3(BATCH*HEADS*(CTX/128)), blk, 0, stream>>>(qbuf, kbuf, vbuf, attnb);

  gemm_f16<1><<<dim3(EMB/128, NROW/128), blk, 0, stream>>>(
      attnb, wprojh, NROW, EMB, EMB,
      b_proj, x, out, nullptr, nullptr, nullptr, nullptr);

  ln_half<<<dim3(NROW), blk, 0, stream>>>(out, ln2g, ln2b, h2buf);

  gemm_f16<2><<<dim3(FF/128, NROW/128), blk, 0, stream>>>(
      h2buf, w1h, NROW, FF, EMB,
      b1, nullptr, nullptr, ff1, nullptr, nullptr, nullptr);

  gemm_f16<1><<<dim3(EMB/128, NROW/128), blk, 0, stream>>>(
      ff1, w2h, NROW, EMB, FF,
      b2, out, out, nullptr, nullptr, nullptr, nullptr);

  (void)in_sizes; (void)n_in; (void)out_size; (void)ws_size;
}
